// ObjWise_10806137716859
// MI455X (gfx1250) — matrix operand order repaired and checked by reading, not compiled, not run on hardware
//
#include <hip/hip_runtime.h>
#include <hip/hip_bf16.h>
#include <stdint.h>

typedef __attribute__((ext_vector_type(16))) __bf16 v16bf;
typedef __attribute__((ext_vector_type(8)))  float  v8f;
typedef __attribute__((ext_vector_type(4)))  float  f32x4;   // native vector for NT loads

#define D_DIM 256
#define TILES_PER_BLOCK 8
#define RS 544  // LDS row stride in bytes: 256*2 + 32 pad (breaks bank conflicts)

// Single-instruction packed fp32->bf16 convert (RNE). One VOP3 op replaces ~5
// integer ops per pair; used 128x for the W register cache + per-tile staging.
__device__ __forceinline__ unsigned bf16pk(float a, float b) {
  unsigned r;
  asm("v_cvt_pk_bf16_f32 %0, %1, %2" : "=v"(r) : "v"(a), "v"(b));
  return r;
}

union BFrag { v16bf v; unsigned u[8]; uint4 q[2]; };

__global__ __launch_bounds__(256)
void objwise_gemm_bf16(const float* __restrict__ x,
                       const uint8_t* __restrict__ mask,
                       const float* __restrict__ W,
                       const float* __restrict__ bias,
                       float* __restrict__ out)
{
  __shared__ __align__(16) unsigned char ldsA[2][16 * RS];

  const int tid  = threadIdx.x;
  const int lane = tid & 31;
  const int wv   = tid >> 5;        // wave 0..7 -> owns 32 output columns
  const int c    = lane & 15;       // column-within-N-tile == row-within-A-tile
  const int g    = lane >> 4;       // lane group (K-half selector)
  const int colBase = wv * 32;

  const long row0 = (long)blockIdx.x * (TILES_PER_BLOCK * 16);

  // ---- Preload W as bf16 B-fragments into registers (L2-resident reads) ----
  // B 32x16 bf16 k order used here (matches A): lane = column (c), lanes 0-15 hold K=0..15,
  // lanes 16-31 hold K=16..31, 2 K-values per VGPR.
  BFrag Bf[8][2];
#pragma unroll
  for (int kt = 0; kt < 8; ++kt) {
#pragma unroll
    for (int j = 0; j < 2; ++j) {
      const int col = colBase + j * 16 + c;
      const int k0  = kt * 32 + g * 16;
      const float* wp = W + (size_t)k0 * D_DIM + col;
#pragma unroll
      for (int p = 0; p < 8; ++p) {
        float w0 = wp[(2 * p)     * D_DIM];
        float w1 = wp[(2 * p + 1) * D_DIM];
        Bf[kt][j].u[p] = bf16pk(w0, w1);
      }
    }
  }

  const float b0 = bias[colBase + c];
  const float b1 = bias[colBase + 16 + c];

  // staging geometry: thread -> (row-in-tile, 16-float K chunk)
  const int trow = tid >> 4;
  const int tk   = (tid & 15) * 16;

  // ---- preload tile 0 into LDS buffer 0 (non-temporal: pure streaming) ----
  f32x4 s[4];
  {
    const f32x4* gp = (const f32x4*)(x + (size_t)(row0 + trow) * D_DIM + tk);
#pragma unroll
    for (int q = 0; q < 4; ++q) s[q] = __builtin_nontemporal_load(gp + q);
    unsigned u[8];
#pragma unroll
    for (int q = 0; q < 4; ++q) {
      u[2 * q]     = bf16pk(s[q].x, s[q].y);
      u[2 * q + 1] = bf16pk(s[q].z, s[q].w);
    }
    uint4* dp = (uint4*)&ldsA[0][trow * RS + tk * 2];
    dp[0] = make_uint4(u[0], u[1], u[2], u[3]);
    dp[1] = make_uint4(u[4], u[5], u[6], u[7]);
  }
  __syncthreads();

#pragma unroll 1
  for (int t = 0; t < TILES_PER_BLOCK; ++t) {
    const int  buf  = t & 1;
    const long rowT = row0 + (long)t * 16;

    // Mask bytes for this tile's 16 rows via WAVE-UNIFORM loads -> SGPRs
    // (tracked by KMcnt, so the wait does not drain the vector prefetch).
    const unsigned long long* mqp = (const unsigned long long*)(mask + rowT);
    const unsigned long long m01 = mqp[0];   // rows 0..7
    const unsigned long long m23 = mqp[1];   // rows 8..15
    const unsigned long long mm  = g ? m23 : m01;

    // issue next tile's global loads early (overlap with WMMA burst)
    if (t + 1 < TILES_PER_BLOCK) {
      const f32x4* gp = (const f32x4*)(x + (size_t)(rowT + 16 + trow) * D_DIM + tk);
#pragma unroll
      for (int q = 0; q < 4; ++q) s[q] = __builtin_nontemporal_load(gp + q);
    }

    // ---- 16x32 output tile per wave: 8 K-steps x 2 N-tiles = 16 WMMAs ----
    v8f acc0 = {}; v8f acc1 = {};
    const unsigned char* abase = &ldsA[buf][c * RS];
#pragma unroll
    for (int kt = 0; kt < 8; ++kt) {
      // A 16x32 bf16 k order used here (matches B): lane holds row c; lanes 0-15: K {0..15};
      // lanes 16-31: K {16..31} -> two contiguous 16B LDS reads.
      BFrag a;
      a.q[0] = *(const uint4*)(abase + kt * 64 + g * 32);
      a.q[1] = *(const uint4*)(abase + kt * 64 + g * 32 + 16);
      acc0 = __builtin_amdgcn_wmma_f32_16x16x32_bf16(
          false, a.v, false, Bf[kt][0].v, (short)0, acc0, false, false);
      acc1 = __builtin_amdgcn_wmma_f32_16x16x32_bf16(
          false, a.v, false, Bf[kt][1].v, (short)0, acc1, false, false);
    }

    // ---- epilogue: bias, relu, token mask, non-temporal fp32 store ----
    // C/D layout: lane col = c (+16 for acc1), VGPR i -> row g*8 + i
    float* op = out + (size_t)(rowT + g * 8) * D_DIM + colBase + c;
#pragma unroll
    for (int i = 0; i < 8; ++i) {
      const bool keep = ((mm >> (8 * i)) & 0xFFull) != 0ull;
      float v0 = fmaxf(acc0[i] + b0, 0.0f);
      float v1 = fmaxf(acc1[i] + b1, 0.0f);
      __builtin_nontemporal_store(keep ? v0 : 0.0f, op + (size_t)i * D_DIM);
      __builtin_nontemporal_store(keep ? v1 : 0.0f, op + (size_t)i * D_DIM + 16);
    }

    // ---- convert & park next tile into the other LDS buffer ----
    if (t + 1 < TILES_PER_BLOCK) {
      unsigned u[8];
#pragma unroll
      for (int q = 0; q < 4; ++q) {
        u[2 * q]     = bf16pk(s[q].x, s[q].y);
        u[2 * q + 1] = bf16pk(s[q].z, s[q].w);
      }
      uint4* dp = (uint4*)&ldsA[buf ^ 1][trow * RS + tk * 2];
      dp[0] = make_uint4(u[0], u[1], u[2], u[3]);
      dp[1] = make_uint4(u[4], u[5], u[6], u[7]);
    }
    __syncthreads();
  }
}

extern "C" void kernel_launch(void* const* d_in, const int* in_sizes, int n_in,
                              void* d_out, int out_size, void* d_ws, size_t ws_size,
                              hipStream_t stream) {
  const float*   x    = (const float*)d_in[0];
  const uint8_t* mask = (const uint8_t*)d_in[1];   // jnp.bool_ -> 1 byte/elem
  const float*   W    = (const float*)d_in[2];
  const float*   bias = (const float*)d_in[3];
  float*         out  = (float*)d_out;

  // 131072 rows / (8 tiles * 16 rows) = 1024 blocks, 256 threads (8 waves)
  hipLaunchKernelGGL(objwise_gemm_bf16, dim3(1024), dim3(256), 0, stream,
                     x, mask, W, bias, out);
}
